// past_decoder_9165460210139
// MI455X (gfx1250) — compile-verified, matrix operand order checked by reading, not run on hardware
//
#include <hip/hip_runtime.h>
#include <hip/hip_bf16.h>

#define OBS_LEN 20
#define NPED    4096
#define HID     1024
#define FOURH   4096

typedef __attribute__((ext_vector_type(16))) __bf16 bf16x16;
typedef __attribute__((ext_vector_type(8)))  __bf16 bf16x8;
typedef __attribute__((ext_vector_type(8)))  float  f32x8;

union Frag { bf16x16 v; bf16x8 h[2]; };

// Tile configuration
#define BM 64          // rows of h per block
#define BN 64          // columns per gate per block
#define BK 64          // K chunk staged in LDS
#define LDA 88         // padded LDS row stride in bf16 elems (176B: 16B-aligned, bank-conflict-free)
#define SMEM_BYTES (4 * 64 * 65 * 4)   // gate exchange buffer dominates (66560 B > A/B staging 56320 B)

__device__ __forceinline__ float sigmoidf(float x) {
    return 1.0f / (1.0f + __expf(-x));
}

__global__ void cvt_f32_to_bf16(const float* __restrict__ src,
                                __bf16* __restrict__ dst, int n) {
    int i = blockIdx.x * blockDim.x + threadIdx.x;
    if (i < n) dst[i] = (__bf16)src[i];
}

__global__ void zero_f32(float* __restrict__ p, int n) {
    int i = blockIdx.x * blockDim.x + threadIdx.x;
    if (i < n) p[i] = 0.0f;
}

__global__ void init_out(float* __restrict__ out, const float* __restrict__ b_out, int n) {
    int i = blockIdx.x * blockDim.x + threadIdx.x;
    if (i < n) out[i] = b_out[i & 1];
}

__global__ __launch_bounds__(256)
void lstm_step(const __bf16* __restrict__ Hin,    // [NPED][HID] bf16
               __bf16* __restrict__ Hout,         // [NPED][HID] bf16 (next step)
               float* __restrict__ Cst,           // [NPED][HID] f32 cell state
               const __bf16* __restrict__ Whh,    // [4H][HID] bf16
               const float* __restrict__ X,       // [NPED][2]  x_t
               const float* __restrict__ Wih,     // [4H][2]
               const float* __restrict__ bih,     // [4H]
               const float* __restrict__ bhh,     // [4H]
               const float* __restrict__ Wout,    // [2][HID]
               float* __restrict__ outT)          // [NPED][2] (pre-init with b_out)
{
    extern __shared__ __align__(16) char smem_raw[];
    __bf16* sA = (__bf16*)smem_raw;           // [BM][LDA]      h tile
    __bf16* sB = sA + BM * LDA;               // [4][BN][LDA]   W_hh tiles (4 gates)
    float*  sG = (float*)smem_raw;            // [4][BM][65]    gate exchange (reuses LDS)

    const int tid    = threadIdx.x;
    const int lane   = tid & 31;
    const int wv     = tid >> 5;       // wave 0..7
    const int g      = wv & 3;         // gate handled by this wave
    const int mhalf  = wv >> 2;        // 0..1: which 32-row half
    const int lrow   = lane & 15;
    const int laneHi = lane >> 4;

    const int m0 = blockIdx.x * BM;
    const int n0 = blockIdx.y * BN;

    f32x8 acc[2][4] = {};

    for (int kc = 0; kc < HID; kc += BK) {
        // ---- global -> LDS staging (16B chunks) ----
        #pragma unroll
        for (int i = 0; i < 2; ++i) {          // A: 64x64 bf16 = 512 chunks
            int c = tid + i * 256;
            int row = c >> 3, col = (c & 7) * 8;
            *(uint4*)(sA + row * LDA + col) =
                *(const uint4*)(Hin + (size_t)(m0 + row) * HID + kc + col);
        }
        #pragma unroll
        for (int i = 0; i < 8; ++i) {          // B: 4 x 64x64 bf16 = 2048 chunks
            int c = tid + i * 256;
            int gg = c >> 9, rr = (c >> 3) & 63, col = (c & 7) * 8;
            *(uint4*)(sB + ((gg << 6) + rr) * LDA + col) =
                *(const uint4*)(Whh + (size_t)(gg * HID + n0 + rr) * HID + kc + col);
        }
        __syncthreads();

        // ---- WMMA over this K chunk (two K=32 sub-steps) ----
        #pragma unroll
        for (int ks = 0; ks < BK; ks += 32) {
            Frag fa[2], fb[4];
            #pragma unroll
            for (int tm = 0; tm < 2; ++tm) {   // A fragments: lane-interleaved 16-bit layout
                const __bf16* p = sA + (mhalf * 32 + tm * 16 + lrow) * LDA + ks + laneHi * 8;
                fa[tm].h[0] = *(const bf16x8*)(p);
                fa[tm].h[1] = *(const bf16x8*)(p + 16);
            }
            #pragma unroll
            for (int tn = 0; tn < 4; ++tn) {   // B fragments (B = Whh^T -> load Whh rows)
                const __bf16* p = sB + ((g << 6) + tn * 16 + lrow) * LDA + ks + laneHi * 8;
                fb[tn].h[0] = *(const bf16x8*)(p);
                fb[tn].h[1] = *(const bf16x8*)(p + 16);
            }
            #pragma unroll
            for (int tm = 0; tm < 2; ++tm)
                #pragma unroll
                for (int tn = 0; tn < 4; ++tn)
                    acc[tm][tn] = __builtin_amdgcn_wmma_f32_16x16x32_bf16(
                        false, fa[tm].v, false, fb[tn].v,
                        (short)0, acc[tm][tn], false, false);
        }
        __syncthreads();
    }

    // ---- exchange gate tiles through LDS ----
    #pragma unroll
    for (int tm = 0; tm < 2; ++tm)
        #pragma unroll
        for (int tn = 0; tn < 4; ++tn)
            #pragma unroll
            for (int e = 0; e < 8; ++e) {
                int mr = mhalf * 32 + tm * 16 + laneHi * 8 + e;  // C layout: M=(lane/16)*8+e
                int nc = tn * 16 + lrow;                          //           N=lane%16
                sG[((g << 6) + mr) * 65 + nc] = acc[tm][tn][e];
            }
    __syncthreads();

    // ---- fused LSTM elementwise + x@W_ih^T + output projection ----
    const int m  = tid >> 2;           // 0..63 local row
    const int nq = tid & 3;            // quad within row
    const int row_g = m0 + m;
    const float x0 = X[row_g * 2 + 0];
    const float x1 = X[row_g * 2 + 1];
    float o0 = 0.0f, o1 = 0.0f;

    #pragma unroll
    for (int k = 0; k < 16; ++k) {
        int n  = nq * 16 + k;
        int jg = n0 + n;
        float gates[4];
        #pragma unroll
        for (int q = 0; q < 4; ++q) {
            int colq = q * HID + jg;
            gates[q] = sG[((q << 6) + m) * 65 + n]
                     + x0 * Wih[colq * 2 + 0] + x1 * Wih[colq * 2 + 1]
                     + bih[colq] + bhh[colq];
        }
        size_t idx = (size_t)row_g * HID + jg;
        float cold = Cst[idx];
        float cnew = sigmoidf(gates[1]) * cold + sigmoidf(gates[0]) * tanhf(gates[2]);
        float hv   = sigmoidf(gates[3]) * tanhf(cnew);
        Cst[idx]  = cnew;
        Hout[idx] = (__bf16)hv;
        o0 += hv * Wout[jg];
        o1 += hv * Wout[HID + jg];
    }
    // reduce the 4 threads of each row, then one atomic per row per column-tile
    o0 += __shfl_down(o0, 2, 4);  o0 += __shfl_down(o0, 1, 4);
    o1 += __shfl_down(o1, 2, 4);  o1 += __shfl_down(o1, 1, 4);
    if (nq == 0) {
        atomicAdd(&outT[row_g * 2 + 0], o0);
        atomicAdd(&outT[row_g * 2 + 1], o1);
    }
}

extern "C" void kernel_launch(void* const* d_in, const int* in_sizes, int n_in,
                              void* d_out, int out_size, void* d_ws, size_t ws_size,
                              hipStream_t stream) {
    (void)in_sizes; (void)n_in; (void)ws_size;

    const float* obs  = (const float*)d_in[0];   // [20][4096][2]
    const float* h0f  = (const float*)d_in[1];   // [4096][1024]
    const float* Wih  = (const float*)d_in[2];   // [4096][2]
    const float* WhhF = (const float*)d_in[3];   // [4096][1024]
    const float* bih  = (const float*)d_in[4];   // [4096]
    const float* bhh  = (const float*)d_in[5];   // [4096]
    const float* Wout = (const float*)d_in[6];   // [2][1024]
    const float* bout = (const float*)d_in[7];   // [2]
    float* out = (float*)d_out;                  // [20][4096][2]

    char* ws = (char*)d_ws;
    __bf16* whh_bf = (__bf16*)ws;                                  //  8 MB
    __bf16* hbuf0  = (__bf16*)(ws + (size_t) 8 * 1024 * 1024);     //  8 MB
    __bf16* hbuf1  = (__bf16*)(ws + (size_t)16 * 1024 * 1024);     //  8 MB
    float*  cst    = (float*) (ws + (size_t)24 * 1024 * 1024);     // 16 MB

    const int HW = FOURH * HID;   // 4M
    const int NH = NPED * HID;    // 4M
    cvt_f32_to_bf16<<<(HW + 255) / 256, 256, 0, stream>>>(WhhF, whh_bf, HW);
    cvt_f32_to_bf16<<<(NH + 255) / 256, 256, 0, stream>>>(h0f, hbuf0, NH);
    zero_f32<<<(NH + 255) / 256, 256, 0, stream>>>(cst, NH);
    init_out<<<(out_size + 255) / 256, 256, 0, stream>>>(out, bout, out_size);

    __bf16* hin = hbuf0;
    __bf16* hout = hbuf1;
    for (int t = 0; t < OBS_LEN; ++t) {
        int tsrc = (t == 0) ? 0 : t - 1;   // step 0 consumes obs[0]; step i consumes obs[i-1]
        lstm_step<<<dim3(NPED / BM, HID / BN), 256, SMEM_BYTES, stream>>>(
            hin, hout, cst, whh_bf,
            obs + (size_t)tsrc * NPED * 2,
            Wih, bih, bhh, Wout,
            out + (size_t)t * NPED * 2);
        __bf16* tmp = hin; hin = hout; hout = tmp;
    }
}
